// PlanStructuredNetwork_22230750724709
// MI455X (gfx1250) — matrix operand order repaired and checked by reading, not compiled, not run on hardware
//
#include <hip/hip_runtime.h>

typedef __attribute__((ext_vector_type(16))) _Float16 v16h;
typedef __attribute__((ext_vector_type(8)))  _Float16 v8h;
typedef __attribute__((ext_vector_type(8)))  float    v8f;
typedef _Float16 half_t;

#define B_SZ    2048
#define NLEAF   1024
#define F_DIM   32
#define H_DIM   128
#define O_DIM   32

// ---- dynamic LDS layout (bytes) ----
#define LDS_CURA  0
#define LDS_CURB  65536
#define LDS_WSW   (65536 + 32768)
#define LDS_BIAS  (LDS_WSW + 65536)
#define LDS_STAGE (LDS_BIAS + 2304)
#define LDS_TOTAL (LDS_STAGE + 131072)   // 297216 B < 320 KB WGP limit

// weight fragment offsets (halfs) inside wsw
//  leaf: L1 @0 (KC1,NT8)=4096, L2 @4096 (KC4,NT8)=16384, L3 @20480 (KC4,NT2)=4096
//  join: L1 @0 (KC3,NT8)=12288, L2 @12288 (KC4,NT8)=16384, L3 @28672 (KC4,NT2)=4096
// bias offsets (floats): sb1 0, sb2 128, sb3 256, jb1 288, jb2 416, jb3 544

union Frag16 { v16h v; uint4 q[2]; };
union Pk4    { half_t h[4]; uint2 u; };
union Pk8    { half_t h[8]; v8h hv; uint4 q; };

// D = A(16xK, weights^T) x B(Kx16, activations) + C   ->  D[channel][batchrow]
__device__ __forceinline__ v8f wmma_f16(v16h a, v16h b, v8f c) {
  return __builtin_amdgcn_wmma_f32_16x16x32_f16(false, a, false, b, (short)0, c,
                                                false, false);
}

// Weight A-fragment: pre-swizzled in LDS as [chunk][lane][16 halfs] contiguous
__device__ __forceinline__ v16h load_wa_frag(const half_t* frag) {
  const int lane = threadIdx.x & 31;
  const half_t* p = frag + lane * 16;
  Frag16 f;
  f.q[0] = *reinterpret_cast<const uint4*>(p);
  f.q[1] = *reinterpret_cast<const uint4*>(p + 8);
  return f.v;
}

// Activation B-fragment from row-major staging (ld = LDK halfs), K-chunk kc:
// lane holds column = batch row (lane&15), K = kc*32 + 16*(lane>=16) + {0..15} contiguous
template<int LDK>
__device__ __forceinline__ v16h load_x_bfrag(const half_t* stage, int kc) {
  const int lane = threadIdx.x & 31;
  const int m   = lane & 15;
  const int kb2 = (lane >> 4) << 4;          // 0 or 16
  const half_t* p = stage + m * LDK + kc * 32 + kb2;
  Frag16 f;
  f.q[0] = *reinterpret_cast<const uint4*>(p);
  f.q[1] = *reinterpret_cast<const uint4*>(p + 8);
  return f.v;
}

// Swizzle fp32 weights W[K][N] into the A-fragment layout of W^T tiles:
// row of A = output channel c = nt*16 + (lane&15);
// K(j) = kc*32 + 16*(lane>=16) + j
__device__ __forceinline__ void load_w_swizzledA(const float* __restrict__ Wg,
                                                 half_t* dst, int KC, int NT, int Ncols) {
  const int total = KC * NT * 512;
  for (int idx = threadIdx.x; idx < total; idx += 256) {
    const int j    = idx & 15;
    const int ln   = (idx >> 4) & 31;
    const int chnk = idx >> 9;
    const int kc   = chnk / NT;
    const int nt   = chnk - kc * NT;
    const int K    = kc * 32 + ((ln >> 4) << 4) + j;
    const int c    = nt * 16 + (ln & 15);
    dst[idx] = (half_t)Wg[K * Ncols + c];
  }
}

// ---------------- staging helpers (float4 loads -> packed f16 stores) -------------

__device__ __forceinline__ void stage_leaf(half_t* S0, const float* __restrict__ g) {
  const int lane = threadIdx.x & 31;
  for (int i = lane; i < 128; i += 32) {          // 128 float4 groups = 16x32
    float4 f = reinterpret_cast<const float4*>(g)[i];
    Pk4 t;
    t.h[0] = (half_t)f.x; t.h[1] = (half_t)f.y;
    t.h[2] = (half_t)f.z; t.h[3] = (half_t)f.w;
    *reinterpret_cast<uint2*>(S0 + i * 4) = t.u;
  }
}

// Stage one join tile: rows = nodes i0..i0+nValid-1; S0 is 16x96 row-major.
__device__ __forceinline__ void stage_join(half_t* S0, const float* __restrict__ fg,
                                           const half_t* pc, int i0, int nValid) {
  const int lane = threadIdx.x & 31;
  if (nValid < 16) {                               // zero-pad (16*96 halfs = 192 uint4)
    uint4 z; z.x = 0; z.y = 0; z.z = 0; z.w = 0;
    for (int i = lane; i < 192; i += 32) reinterpret_cast<uint4*>(S0)[i] = z;
  }
  // cols 0..31: internal feats fp32 -> f16 (8 float4 groups per row)
  for (int i = lane; i < nValid * 8; i += 32) {
    const int m = i >> 3, c = (i & 7) * 4;
    float4 f = *reinterpret_cast<const float4*>(fg + m * 32 + c);
    Pk4 t;
    t.h[0] = (half_t)f.x; t.h[1] = (half_t)f.y;
    t.h[2] = (half_t)f.z; t.h[3] = (half_t)f.w;
    *reinterpret_cast<uint2*>(S0 + m * 96 + c) = t.u;
  }
  // cols 32..95: children = prev rows 2i,2i+1 (64 contiguous halfs = 8 uint4)
  for (int i = lane; i < nValid * 8; i += 32) {
    const int m = i >> 3, p = i & 7;
    reinterpret_cast<uint4*>(S0 + m * 96 + 32)[p] =
        reinterpret_cast<const uint4*>(pc + (size_t)(i0 + m) * 64)[p];
  }
}

// Bias-initialized accumulator: acc[v] = biasL[nt*16 + co + v]
__device__ __forceinline__ v8f bias_acc(const float* biasL, int nt, int co) {
  float4 b0 = *reinterpret_cast<const float4*>(biasL + nt * 16 + co);
  float4 b1 = *reinterpret_cast<const float4*>(biasL + nt * 16 + co + 4);
  v8f acc;
  acc[0] = b0.x; acc[1] = b0.y; acc[2] = b0.z; acc[3] = b0.w;
  acc[4] = b1.x; acc[5] = b1.y; acc[6] = b1.z; acc[7] = b1.w;
  return acc;
}

// ReLU applied AFTER f16 conversion: cvt results are canonical, so the packed
// f16 maxnum needs no canonicalize ops (4x v_cvt_pk_f16_f32 + 4x v_pk_max_num_f16)
__device__ __forceinline__ void store_relu_pk(half_t* dst, v8f acc) {
  Pk8 t;
#pragma unroll
  for (int v = 0; v < 8; ++v) t.h[v] = (half_t)acc[v];
  v8h z = {};
  t.hv = __builtin_elementwise_max(t.hv, z);
  *reinterpret_cast<uint4*>(dst) = t.q;
}

__device__ __forceinline__ void store_pk(half_t* dst, v8f acc) {
  Pk8 t;
#pragma unroll
  for (int v = 0; v < 8; ++v) t.h[v] = (half_t)acc[v];
  *reinterpret_cast<uint4*>(dst) = t.q;
}

// ---------------- MLP layers (transposed GEMM: D = W^T x X) -----------------------

template<int LDK, int KC, int NT>
__device__ __forceinline__ void mlp_hidden1(const half_t* X0, const half_t* wfrag,
                                            const float* biasL, half_t* H0) {
  const int lane = threadIdx.x & 31;
  const int m  = lane & 15;                 // batch row
  const int co = (lane >> 4) << 3;          // channel offset within N-tile
  v16h bx[KC];
#pragma unroll
  for (int kc = 0; kc < KC; ++kc) bx[kc] = load_x_bfrag<LDK>(X0, kc);
#pragma unroll
  for (int nt = 0; nt < NT; ++nt) {
    v8f acc = bias_acc(biasL, nt, co);
#pragma unroll
    for (int kc = 0; kc < KC; ++kc)
      acc = wmma_f16(load_wa_frag(wfrag + (kc * NT + nt) * 512), bx[kc], acc);
    store_relu_pk(H0 + m * 128 + nt * 16 + co, acc);
  }
}

// 2x2 tile variant: 2 M-tiles x 2 N-tiles = 4 independent WMMAs per K-chunk
// (shared operands, and enough independent WMMAs to hide the WMMA->VALU hazard)
template<int LDK, int KC, int NT>
__device__ __forceinline__ void mlp_hidden_pair(const half_t* X0, const half_t* X1,
                                                const half_t* wfrag, const float* biasL,
                                                half_t* H0, half_t* H1) {
  const int lane = threadIdx.x & 31;
  const int m  = lane & 15;
  const int co = (lane >> 4) << 3;
  v16h bx0[KC], bx1[KC];
#pragma unroll
  for (int kc = 0; kc < KC; ++kc) {
    bx0[kc] = load_x_bfrag<LDK>(X0, kc);
    bx1[kc] = load_x_bfrag<LDK>(X1, kc);
  }
#pragma unroll
  for (int nt = 0; nt < NT; nt += 2) {
    v8f a00 = bias_acc(biasL, nt, co);
    v8f a01 = a00;
    v8f a10 = bias_acc(biasL, nt + 1, co);
    v8f a11 = a10;
#pragma unroll
    for (int kc = 0; kc < KC; ++kc) {
      v16h aw0 = load_wa_frag(wfrag + (kc * NT + nt) * 512);
      v16h aw1 = load_wa_frag(wfrag + (kc * NT + nt + 1) * 512);
      a00 = wmma_f16(aw0, bx0[kc], a00);
      a01 = wmma_f16(aw0, bx1[kc], a01);
      a10 = wmma_f16(aw1, bx0[kc], a10);
      a11 = wmma_f16(aw1, bx1[kc], a11);
    }
    store_relu_pk(H0 + m * 128 + nt * 16 + co,       a00);
    store_relu_pk(H1 + m * 128 + nt * 16 + co,       a01);
    store_relu_pk(H0 + m * 128 + (nt + 1) * 16 + co, a10);
    store_relu_pk(H1 + m * 128 + (nt + 1) * 16 + co, a11);
  }
}

// Output layer (K=128, N=32): bias in acc init, no relu, f16 rows -> cur buffer
__device__ __forceinline__ void mlp_out1(const half_t* X0, const half_t* wfrag,
                                         const float* biasL, half_t* cur,
                                         int row0, int nValid) {
  const int lane = threadIdx.x & 31;
  const int m  = lane & 15;
  const int co = (lane >> 4) << 3;
  v16h bx[4];
#pragma unroll
  for (int kc = 0; kc < 4; ++kc) bx[kc] = load_x_bfrag<128>(X0, kc);
#pragma unroll
  for (int nt = 0; nt < 2; ++nt) {
    v8f acc = bias_acc(biasL, nt, co);
#pragma unroll
    for (int kc = 0; kc < 4; ++kc)
      acc = wmma_f16(load_wa_frag(wfrag + (kc * 2 + nt) * 512), bx[kc], acc);
    Pk8 t;
#pragma unroll
    for (int v = 0; v < 8; ++v) t.h[v] = (half_t)acc[v];
    if (m < nValid)
      *reinterpret_cast<uint4*>(cur + (row0 + m) * 32 + nt * 16 + co) = t.q;
  }
}

__device__ __forceinline__ void mlp_out_pair(const half_t* X0, const half_t* X1,
                                             const half_t* wfrag, const float* biasL,
                                             half_t* cur, int row0) {   // 32 valid rows
  const int lane = threadIdx.x & 31;
  const int m  = lane & 15;
  const int co = (lane >> 4) << 3;
  v16h bx0[4], bx1[4];
#pragma unroll
  for (int kc = 0; kc < 4; ++kc) {
    bx0[kc] = load_x_bfrag<128>(X0, kc);
    bx1[kc] = load_x_bfrag<128>(X1, kc);
  }
  v8f a00 = bias_acc(biasL, 0, co);
  v8f a01 = a00;
  v8f a10 = bias_acc(biasL, 1, co);
  v8f a11 = a10;
#pragma unroll
  for (int kc = 0; kc < 4; ++kc) {
    v16h aw0 = load_wa_frag(wfrag + (kc * 2) * 512);
    v16h aw1 = load_wa_frag(wfrag + (kc * 2 + 1) * 512);
    a00 = wmma_f16(aw0, bx0[kc], a00);
    a01 = wmma_f16(aw0, bx1[kc], a01);
    a10 = wmma_f16(aw1, bx0[kc], a10);
    a11 = wmma_f16(aw1, bx1[kc], a11);
  }
  store_pk(cur + (row0 + m) * 32 + co,           a00);
  store_pk(cur + (row0 + 16 + m) * 32 + co,      a01);
  store_pk(cur + (row0 + m) * 32 + 16 + co,      a10);
  store_pk(cur + (row0 + 16 + m) * 32 + 16 + co, a11);
}

// Final node: only (row 0, channel 0) needed -> lane 0 (m=0,co=0), acc[0], nt=0
__device__ __forceinline__ void mlp_out_final(const half_t* X0, const half_t* wfrag,
                                              const float* biasL, float* outp) {
  const int lane = threadIdx.x & 31;
  const int co = (lane >> 4) << 3;
  v16h bx[4];
#pragma unroll
  for (int kc = 0; kc < 4; ++kc) bx[kc] = load_x_bfrag<128>(X0, kc);
  v8f acc = bias_acc(biasL, 0, co);
#pragma unroll
  for (int kc = 0; kc < 4; ++kc)
    acc = wmma_f16(load_wa_frag(wfrag + (kc * 2) * 512), bx[kc], acc);
  if (threadIdx.x == 0) *outp = acc[0];
}

__global__ __launch_bounds__(256, 1)
void tree_mlp_kernel(const float* __restrict__ leaf_feats,
                     const float* __restrict__ internal_feats,
                     const float* __restrict__ sW1, const float* __restrict__ sb1,
                     const float* __restrict__ sW2, const float* __restrict__ sb2,
                     const float* __restrict__ sW3, const float* __restrict__ sb3,
                     const float* __restrict__ jW1, const float* __restrict__ jb1,
                     const float* __restrict__ jW2, const float* __restrict__ jb2,
                     const float* __restrict__ jW3, const float* __restrict__ jb3,
                     float* __restrict__ out) {
  extern __shared__ char lds[];
  half_t* curA  = (half_t*)(lds + LDS_CURA);
  half_t* curB  = (half_t*)(lds + LDS_CURB);
  half_t* wsw   = (half_t*)(lds + LDS_WSW);
  float*  bias  = (float*)(lds + LDS_BIAS);
  half_t* stage = (half_t*)(lds + LDS_STAGE);

  const int tid  = threadIdx.x;
  const int wave = tid >> 5;
  const int b    = blockIdx.x;

  half_t* S0a = stage + wave * 8192;   // 4 buffers x 2048 halfs per wave
  half_t* S1a = S0a + 2048;
  half_t* S0b = S0a + 4096;
  half_t* S1b = S0a + 6144;

  // ---- biases (both MLPs) ----
  for (int i = tid; i < H_DIM; i += 256) {
    bias[i]       = sb1[i];
    bias[128 + i] = sb2[i];
    bias[288 + i] = jb1[i];
    bias[416 + i] = jb2[i];
  }
  for (int i = tid; i < O_DIM; i += 256) {
    bias[256 + i] = sb3[i];
    bias[544 + i] = jb3[i];
  }
  // ---- leaf weights, swizzled as W^T A-fragments ----
  load_w_swizzledA(sW1, wsw + 0,     1, 8, 128);
  load_w_swizzledA(sW2, wsw + 4096,  4, 8, 128);
  load_w_swizzledA(sW3, wsw + 20480, 4, 2, 32);
  __syncthreads();

  // ================= leaf phase: 32 pairs of 16-row tiles =================
  const float* lg = leaf_feats + (size_t)b * NLEAF * F_DIM;
  for (int p = wave; p < 32; p += 8) {
    const int l0 = p * 32;
    stage_leaf(S0a, lg + (size_t)l0 * F_DIM);
    stage_leaf(S0b, lg + (size_t)(l0 + 16) * F_DIM);
    mlp_hidden_pair<32, 1, 8>(S0a, S0b, wsw + 0, bias + 0, S1a, S1b);
    mlp_hidden_pair<128, 4, 8>(S1a, S1b, wsw + 4096, bias + 128, S0a, S0b);
    mlp_out_pair(S0a, S0b, wsw + 20480, bias + 256, curA, l0);
  }
  __syncthreads();

  // ---- join weights overwrite leaf weights ----
  load_w_swizzledA(jW1, wsw + 0,     3, 8, 128);
  load_w_swizzledA(jW2, wsw + 12288, 4, 8, 128);
  load_w_swizzledA(jW3, wsw + 28672, 4, 2, 32);
  __syncthreads();

  // ================= tree reduction =================
  const float* ig = internal_feats + (size_t)b * (NLEAF - 1) * F_DIM;
  int off = 0;
  int n   = NLEAF / 2;
  half_t* pc = curA;   // previous level (2n rows of 32 f16)
  half_t* co = curB;   // this level output (n rows of 32 f16)
  while (n >= 1) {
    const int tiles = (n + 15) >> 4;
    if (tiles >= 2) {                        // n in {512,256,128,64,32}: even tiles
      const int pairs = tiles >> 1;
      for (int p = wave; p < pairs; p += 8) {
        const int i0 = p * 32;
        stage_join(S0a, ig + (size_t)(off + i0) * F_DIM, pc, i0, 16);
        stage_join(S0b, ig + (size_t)(off + i0 + 16) * F_DIM, pc, i0 + 16, 16);
        mlp_hidden_pair<96, 3, 8>(S0a, S0b, wsw + 0, bias + 288, S1a, S1b);
        mlp_hidden_pair<128, 4, 8>(S1a, S1b, wsw + 12288, bias + 416, S0a, S0b);
        mlp_out_pair(S0a, S0b, wsw + 28672, bias + 544, co, i0);
      }
    } else if (wave == 0) {                  // n in {16,8,4,2,1}: single tile
      stage_join(S0a, ig + (size_t)off * F_DIM, pc, 0, n);
      mlp_hidden1<96, 3, 8>(S0a, wsw + 0, bias + 288, S1a);
      mlp_hidden1<128, 4, 8>(S1a, wsw + 12288, bias + 416, S0a);
      if (n > 1) mlp_out1(S0a, wsw + 28672, bias + 544, co, 0, n);
      else       mlp_out_final(S0a, wsw + 28672, bias + 544, out + b);
    }
    __syncthreads();
    off += n;
    half_t* tmp = pc; pc = co; co = tmp;
    n >>= 1;
  }
}

extern "C" void kernel_launch(void* const* d_in, const int* in_sizes, int n_in,
                              void* d_out, int out_size, void* d_ws, size_t ws_size,
                              hipStream_t stream) {
  (void)in_sizes; (void)n_in; (void)d_ws; (void)ws_size; (void)out_size;
  const float* leaf = (const float*)d_in[0];
  const float* intf = (const float*)d_in[1];
  const float* sW1  = (const float*)d_in[2];
  const float* sb1  = (const float*)d_in[3];
  const float* sW2  = (const float*)d_in[4];
  const float* sb2  = (const float*)d_in[5];
  const float* sW3  = (const float*)d_in[6];
  const float* sb3  = (const float*)d_in[7];
  const float* jW1  = (const float*)d_in[8];
  const float* jb1  = (const float*)d_in[9];
  const float* jW2  = (const float*)d_in[10];
  const float* jb2  = (const float*)d_in[11];
  const float* jW3  = (const float*)d_in[12];
  const float* jb3  = (const float*)d_in[13];
  float* out = (float*)d_out;

  hipFuncSetAttribute(reinterpret_cast<const void*>(tree_mlp_kernel),
                      hipFuncAttributeMaxDynamicSharedMemorySize, LDS_TOTAL);
  tree_mlp_kernel<<<B_SZ, 256, LDS_TOTAL, stream>>>(
      leaf, intf, sW1, sb1, sW2, sb2, sW3, sb3,
      jW1, jb1, jW2, jb2, jW3, jb3, out);
}
